// WikiHMM_48292612276523
// MI455X (gfx1250) — matrix operand order repaired and checked by reading, not compiled, not run on hardware
//
#include <hip/hip_runtime.h>
#include <hip/hip_bf16.h>

// HMM forward (scaled / rescaled linear-space) + softmax outputs for MI455X (gfx1250).
// Recurrence is restructured from log-space logsumexp into per-step GEMMs so it maps
// onto v_wmma_f32_16x16x32_f16 (wave32 WMMA), with f32 accumulate and per-batch max
// rescaling for numerical stability (classic scaled forward algorithm).
// The transition-matrix (WMMA B) fragments are loop-invariant across all 255 steps,
// so they are hoisted into 128 VGPRs per wave; only alpha fragments stream from LDS.

#define Hh 512
#define Vv 50003
#define Bb 32
#define Tt 256
#define NWG 8
#define TPB 256
#define DIAG_MAXF 0.7f

// d_out layout (floats, concatenated in reference return order)
#define OUT_PI 0
#define OUT_A  512
#define OUT_B  (512 + 512 * 512)            // 262656
#define OUT_SC (OUT_B + 512 * Vv)           // 25864192: [ll_loss, ent, cnct, diag_reg, self_trans]

// d_ws layout (float offsets)
#define WS_LOGPI 0
#define WS_LSE   512
#define WS_ENT   1024
#define WS_CTR   1536          // unsigned barrier counter (64B-aligned region)
#define WS_LA    1552          // two ping-pong alpha buffers, each 32*512 f32

typedef __attribute__((ext_vector_type(16))) _Float16 v16h;
typedef __attribute__((ext_vector_type(8)))  float    v8f;

// ---------------------------------------------------------------- grid barrier
__device__ inline void gridbar(unsigned* ctr, int tid) {
  __syncthreads();
  if (tid == 0) {
    __threadfence();
    unsigned t = __hip_atomic_fetch_add(ctr, 1u, __ATOMIC_ACQ_REL, __HIP_MEMORY_SCOPE_AGENT);
    unsigned tgt = (t / NWG + 1u) * NWG;
    while (__hip_atomic_load(ctr, __ATOMIC_ACQUIRE, __HIP_MEMORY_SCOPE_AGENT) < tgt)
      __builtin_amdgcn_s_sleep(8);
  }
  __syncthreads();
  __threadfence();
}

// ---------------------------------------------------------------- kernel: pi
__global__ void k_pi(const float* __restrict__ init_logits, float* __restrict__ out,
                     float* __restrict__ ws) {
  __shared__ float red[Hh];
  int tid = threadIdx.x;
  float x = init_logits[tid];
  red[tid] = x; __syncthreads();
  for (int s = Hh / 2; s > 0; s >>= 1) { if (tid < s) red[tid] = fmaxf(red[tid], red[tid + s]); __syncthreads(); }
  float mx = red[0]; __syncthreads();
  red[tid] = __expf(x - mx); __syncthreads();
  for (int s = Hh / 2; s > 0; s >>= 1) { if (tid < s) red[tid] += red[tid + s]; __syncthreads(); }
  float lse = mx + logf(red[0]);
  float lp = x - lse;
  ws[WS_LOGPI + tid] = lp;
  out[OUT_PI + tid] = __expf(lp);
  if (tid == 0) *((unsigned*)(ws + WS_CTR)) = 0u;   // reset grid barrier every launch
}

// ---------------------------------------------------------------- kernel: transition softmax + entropy partials
__global__ void k_trans(const float* __restrict__ tl, float* __restrict__ out,
                        float* __restrict__ ws) {
  __shared__ float red[Hh];
  int tid = threadIdx.x, h = blockIdx.x;
  float x = tl[h * Hh + tid];
  red[tid] = x; __syncthreads();
  for (int s = Hh / 2; s > 0; s >>= 1) { if (tid < s) red[tid] = fmaxf(red[tid], red[tid + s]); __syncthreads(); }
  float mx = red[0]; __syncthreads();
  red[tid] = __expf(x - mx); __syncthreads();
  for (int s = Hh / 2; s > 0; s >>= 1) { if (tid < s) red[tid] += red[tid + s]; __syncthreads(); }
  float lse = mx + logf(red[0]); __syncthreads();
  float la = x - lse;
  float a = __expf(la);
  out[OUT_A + h * Hh + tid] = a;
  red[tid] = -a * la; __syncthreads();
  for (int s = Hh / 2; s > 0; s >>= 1) { if (tid < s) red[tid] += red[tid + s]; __syncthreads(); }
  if (tid == 0) ws[WS_ENT + h] = red[0];
}

// ---------------------------------------------------------------- kernel: emission softmax (bandwidth-dominant)
__global__ void k_emis(const float* __restrict__ el, float* __restrict__ out,
                       float* __restrict__ ws) {
  __shared__ float red[256];
  int tid = threadIdx.x, h = blockIdx.x;
  const float* row = el + (size_t)h * Vv;
  float m = -3.4e38f;
  for (int i = tid; i < Vv; i += 256) {
    __builtin_prefetch(row + i + 4096, 0, 0);
    m = fmaxf(m, row[i]);
  }
  red[tid] = m; __syncthreads();
  for (int s = 128; s > 0; s >>= 1) { if (tid < s) red[tid] = fmaxf(red[tid], red[tid + s]); __syncthreads(); }
  m = red[0]; __syncthreads();
  float sum = 0.f;
  for (int i = tid; i < Vv; i += 256) sum += __expf(row[i] - m);
  red[tid] = sum; __syncthreads();
  for (int s = 128; s > 0; s >>= 1) { if (tid < s) red[tid] += red[tid + s]; __syncthreads(); }
  float lse = m + logf(red[0]);
  if (tid == 0) ws[WS_LSE + h] = lse;
  float* orow = out + OUT_B + (size_t)h * Vv;
  for (int i = tid; i < Vv; i += 256) orow[i] = __expf(row[i] - lse);
}

// ---------------------------------------------------------------- kernel: persistent WMMA forward recurrence
// 8 WGs x 256 thr (8 wave32). Each wave owns one 16x16 (M,N) tile of the 32x64
// per-WG output slab. The 512x16 column slice of A needed by a wave is held
// register-resident as 16 pre-swizzled B fragments (128 VGPRs, loop-invariant).
// alpha_hat is rescaled+converted each step into LDS in the ISA A-fragment layout.
__global__ __launch_bounds__(TPB, 1) void k_forward(const int* __restrict__ text,
                                                    const float* __restrict__ emis,
                                                    const float* __restrict__ Aout,
                                                    float* __restrict__ ws) {
  __shared__ _Float16 sAl[Bb * Hh];   // 16384 halves: [mtile][kblock][lane][e] (A-frag order)
  __shared__ float pmax[TPB];
  __shared__ float smax[Bb];
  __shared__ int   stok[Bb];

  int tid = threadIdx.x, wg = blockIdx.x;
  int lane = tid & 31, w = tid >> 5;
  int mtile = w & 1, ntile = w >> 1;            // 2 x 4 tiles of 16x16 cover 32 x 64
  const float* logpi = ws + WS_LOGPI;
  const float* lse   = ws + WS_LSE;
  unsigned* ctr = (unsigned*)(ws + WS_CTR);
  float* la0 = ws + WS_LA;
  float* la1 = la0 + Bb * Hh;

  // Register-resident B fragments: ISA layout lane L: N = L&15, K(e) = (e&7) + (e>=8 ? 16:0) + 8*(L>=16).
  // One-time global load of this wave's 512x16 column slice of A (amortized over 255 steps).
  v16h breg[16];
  {
    int jg0 = wg * 64 + ntile * 16 + (lane & 15);
    int khi = (lane >= 16) ? 16 : 0;
#pragma unroll
    for (int kb5 = 0; kb5 < 16; ++kb5) {
#pragma unroll
      for (int e = 0; e < 16; ++e) {
        int k = kb5 * 32 + (e & 7) + ((e & 8) << 1) + (khi >> 1);
        breg[kb5][e] = (_Float16)Aout[k * Hh + jg0];
      }
    }
  }

  // t = 0 init: log_alpha0[b,j] = log_pi[j] + (emission_logits[j,tok] - lse[j])
  for (int idx = tid; idx < Bb * 64; idx += TPB) {
    int b = idx >> 6, jl = idx & 63, jg = wg * 64 + jl;
    int tok = text[b * Tt + 0];
    la0[b * Hh + jg] = logpi[jg] + emis[(size_t)jg * Vv + tok] - lse[jg];
  }
  gridbar(ctr, tid);

  for (int t = 1; t < Tt; ++t) {
    const float* lin  = (t & 1) ? la0 : la1;
    float*       lout = (t & 1) ? la1 : la0;

    if (tid < Bb) stok[tid] = text[tid * Tt + t];
    // per-batch max (rescaling factor)
    {
      int b = tid >> 3, ch = tid & 7;
      float pm = -3.4e38f;
      const float* r = lin + b * Hh + ch * 64;
      for (int i = 0; i < 64; ++i) pm = fmaxf(pm, r[i]);
      pmax[tid] = pm;
    }
    __syncthreads();
    if (tid < Bb) {
      float m = pmax[tid * 8];
      for (int i = 1; i < 8; ++i) m = fmaxf(m, pmax[tid * 8 + i]);
      smax[tid] = m;
    }
    __syncthreads();
    // alpha_hat = exp(la - max), stored directly in A-fragment layout:
    // lane L: M = L&15, K(e) = e + (e>=8 ? 8:0) + 8*(L>=16)
    for (int idx = tid; idx < Bb * Hh; idx += TPB) {
      int b = idx >> 9, k = idx & (Hh - 1);
      float v = __expf(lin[idx] - smax[b]);
      int kb5 = k >> 5, kr = k & 31;
      int laneHi = (kr >> 3) & 1;
      int e = (kr & 7) | ((kr & 16) >> 1);
      sAl[((((b >> 4) * 16 + kb5) * 32) + laneHi * 16 + (b & 15)) * 16 + e] = (_Float16)v;
    }
    __syncthreads();

    // y = alpha_hat @ A_slice  (16 chained WMMAs over K=512; B operands in VGPRs)
    v8f c = {0.f, 0.f, 0.f, 0.f, 0.f, 0.f, 0.f, 0.f};
    const v16h* pA = (const v16h*)sAl;
#pragma unroll
    for (int kb5 = 0; kb5 < 16; ++kb5) {
      v16h af = pA[(mtile * 16 + kb5) * 32 + lane];
      c = __builtin_amdgcn_wmma_f32_16x16x32_f16(false, af, false, breg[kb5], (short)0, c,
                                                 false, false);
    }

    // epilogue: la' = max + log(y) + (emission_logits[j,tok] - lse[j])
    // C/D layout: VGPR r, lane L -> M = r + 8*(L>=16), N = L&15
    int n  = lane & 15;
    int jg = wg * 64 + ntile * 16 + n;
    float jlse = lse[jg];
    const float* ecol = emis + (size_t)jg * Vv;
#pragma unroll
    for (int r = 0; r < 8; ++r) {
      int b = mtile * 16 + r + ((lane >= 16) ? 8 : 0);
      float y = fmaxf(c[r], 1e-37f);
      lout[b * Hh + jg] = smax[b] + logf(y) + (ecol[stok[b]] - jlse);
    }
    gridbar(ctr, tid);
  }
}

// ---------------------------------------------------------------- kernel: final scalars
__global__ void k_final(const float* __restrict__ Aout, const float* __restrict__ ws,
                        float* __restrict__ out_sc) {
  __shared__ float red[Hh];
  int tid = threadIdx.x;
  const float* laf = ws + WS_LA + Bb * Hh;      // final alphas live in buf1 (T-1 = 255 odd)

  float llsum = 0.f;
  for (int b = 0; b < Bb; ++b) {
    float v = laf[b * Hh + tid];
    red[tid] = v; __syncthreads();
    for (int s = Hh / 2; s > 0; s >>= 1) { if (tid < s) red[tid] = fmaxf(red[tid], red[tid + s]); __syncthreads(); }
    float mx = red[0]; __syncthreads();
    red[tid] = __expf(v - mx); __syncthreads();
    for (int s = Hh / 2; s > 0; s >>= 1) { if (tid < s) red[tid] += red[tid + s]; __syncthreads(); }
    if (tid == 0) llsum += mx + logf(red[0]);
    __syncthreads();
  }
  if (tid == 0) out_sc[0] = -llsum / (float)Bb;

  // transition entropy: mean over rows of per-row sums
  red[tid] = ws[WS_ENT + tid]; __syncthreads();
  for (int s = Hh / 2; s > 0; s >>= 1) { if (tid < s) red[tid] += red[tid + s]; __syncthreads(); }
  if (tid == 0) out_sc[1] = red[0] / (float)Hh;
  __syncthreads();

  // cnct_const from column means of A
  float cs = 0.f;
  for (int i = 0; i < Hh; ++i) cs += Aout[i * Hh + tid];
  float mpc = cs / (float)Hh;
  red[tid] = -mpc * logf(mpc + 1e-10f); __syncthreads();
  for (int s = Hh / 2; s > 0; s >>= 1) { if (tid < s) red[tid] += red[tid + s]; __syncthreads(); }
  if (tid == 0) out_sc[2] = red[0];
  __syncthreads();

  // diag regularizers
  float d  = Aout[tid * Hh + tid];
  float rl = fmaxf(d - DIAG_MAXF, 0.f);
  red[tid] = rl * rl; __syncthreads();
  for (int s = Hh / 2; s > 0; s >>= 1) { if (tid < s) red[tid] += red[tid + s]; __syncthreads(); }
  if (tid == 0) out_sc[3] = red[0];
  __syncthreads();
  red[tid] = d * d; __syncthreads();
  for (int s = Hh / 2; s > 0; s >>= 1) { if (tid < s) red[tid] += red[tid + s]; __syncthreads(); }
  if (tid == 0) out_sc[4] = 100.0f / sqrtf(red[0]);
}

// ---------------------------------------------------------------- launch
extern "C" void kernel_launch(void* const* d_in, const int* in_sizes, int n_in,
                              void* d_out, int out_size, void* d_ws, size_t ws_size,
                              hipStream_t stream) {
  const int*   text   = (const int*)d_in[0];    // (B,T) token ids
  const float* initl  = (const float*)d_in[1];  // (H,)
  const float* transl = (const float*)d_in[2];  // (H,H)
  const float* emisl  = (const float*)d_in[3];  // (H,V)
  float* out = (float*)d_out;
  float* ws  = (float*)d_ws;

  k_pi    <<<1,   Hh,  0, stream>>>(initl, out, ws);
  k_trans <<<Hh,  Hh,  0, stream>>>(transl, out, ws);
  k_emis  <<<Hh,  256, 0, stream>>>(emisl, out, ws);
  k_forward<<<NWG, TPB, 0, stream>>>(text, emisl, out + OUT_A, ws);
  k_final <<<1,   Hh,  0, stream>>>(out + OUT_A, ws, out + OUT_SC);
}
